// Model_37658273251909
// MI455X (gfx1250) — matrix operand order repaired and checked by reading, not compiled, not run on hardware
//
#include <hip/hip_runtime.h>
#include <hip/hip_bf16.h>

typedef __attribute__((ext_vector_type(16))) __bf16 v16bf;
typedef __attribute__((ext_vector_type(8)))  __bf16 v8bf;
typedef __attribute__((ext_vector_type(8)))  float  v8f;

constexpr int kH   = 1024;
constexpr int kB   = 256;
constexpr int kG   = 4096;   // 4*H
constexpr int kTEN = 128;
constexpr int kTDE = 128;
constexpr int kIN  = 4;

// branch-free activations on hardware transcendentals (v_exp_f32 / v_rcp_f32)
__device__ __forceinline__ float fsig(float x) {
  return __builtin_amdgcn_rcpf(1.f + __builtin_amdgcn_exp2f(-1.44269504088896340736f * x));
}
__device__ __forceinline__ float ftanh(float x) {
  return 2.f * __builtin_amdgcn_rcpf(1.f + __builtin_amdgcn_exp2f(-2.88539008177792681472f * x)) - 1.f;
}

// ---------------- setup kernels ----------------

__global__ __launch_bounds__(256) void cvt_f32_bf16(const float* __restrict__ src,
                                                    __bf16* __restrict__ dst, int n) {
  int i = blockIdx.x * 256 + threadIdx.x;
  if (i < n) dst[i] = (__bf16)src[i];
}

__global__ __launch_bounds__(256) void init_state(const float* __restrict__ h0,
                                                  const float* __restrict__ c0,
                                                  __bf16* __restrict__ h_bf,
                                                  float* __restrict__ c) {
  int i = blockIdx.x * 256 + threadIdx.x;   // grid covers exactly B*H
  h_bf[i] = (__bf16)h0[i];
  c[i] = c0[i];
}

// ---------------- fused LSTM step ----------------
// gates = (x@Wih^T + b) seeded into WMMA accumulators, + h@Whh^T via bf16 WMMA,
// then in-block LDS gate exchange + branch-free cell update.
// Block = 8 waves: 2 M-waves (32 batch rows) x 4 gate-waves; wave tile 32x32 of one gate.

__global__ __launch_bounds__(256) void lstm_step(
    const __bf16* __restrict__ W,    // [4096,1024] Whh bf16, row-major
    const float*  __restrict__ Wih,  // [4096,4]
    const float*  __restrict__ bias, // [4096]
    const float*  __restrict__ x_t,  // [256,4]
    const __bf16* __restrict__ h_in, // [256,1024] bf16
    float*        __restrict__ c,    // [256,1024] f32 (in/out)
    __bf16*       __restrict__ h_out)// [256,1024] bf16
{
  __shared__ float lds_g[4][64][32];   // gate pre-activations, 32 KB
  __shared__ float lds_x[64][4];       // x slice for this row tile

  const int tid  = threadIdx.x;
  const int lane = tid & 31;
  const int wave = tid >> 5;
  const int half = lane >> 4;          // lane group 0..15 vs 16..31
  const int l16  = lane & 15;
  const int wm   = wave & 1;           // M-wave (32 rows)
  const int gate = wave >> 1;          // i/f/g/o
  const int m0   = blockIdx.x * 32;    // m tile within H
  const int rowb = blockIdx.y * 64;    // batch-row tile

  // stage the 64x4 x slice (256 elements == 256 threads)
  lds_x[tid >> 2][tid & 3] = x_t[(rowb + (tid >> 2)) * 4 + (tid & 3)];
  __syncthreads();

  // seed accumulators with bias + x @ Wih^T (WMMA accumulates on top)
  v8f acc[2][2];
  #pragma unroll
  for (int nf = 0; nf < 2; ++nf) {
    const int col = gate * kH + m0 + nf * 16 + l16;
    const float w0 = Wih[col * 4 + 0], w1 = Wih[col * 4 + 1];
    const float w2 = Wih[col * 4 + 2], w3 = Wih[col * 4 + 3];
    const float bc = bias[col];
    #pragma unroll
    for (int mf = 0; mf < 2; ++mf)
      #pragma unroll
      for (int r = 0; r < 8; ++r) {
        const int rl = wm * 32 + mf * 16 + half * 8 + r;   // local batch row
        acc[mf][nf][r] = bc + lds_x[rl][0] * w0 + lds_x[rl][1] * w1
                            + lds_x[rl][2] * w2 + lds_x[rl][3] * w3;
      }
  }

  // K loop: h @ Whh^T
  #pragma unroll 2
  for (int kk = 0; kk < kH; kk += 32) {
    v16bf afr[2], bfr[2];
    // A fragment: lane half 0 -> K 0..15, half 1 -> K 16..31 (contiguous)
    #pragma unroll
    for (int mf = 0; mf < 2; ++mf) {
      const __bf16* p = h_in + (rowb + wm * 32 + mf * 16 + l16) * kH + kk + half * 16;
      v8bf lo = *(const v8bf*)p;
      v8bf hi = *(const v8bf*)(p + 8);
      #pragma unroll
      for (int i = 0; i < 8; ++i) { afr[mf][i] = lo[i]; afr[mf][8 + i] = hi[i]; }
    }
    // B fragment: lane = column, lane half selects K 0..15 vs 16..31 (contiguous)
    #pragma unroll
    for (int nf = 0; nf < 2; ++nf) {
      const __bf16* q = W + (gate * kH + m0 + nf * 16 + l16) * kH + kk + half * 16;
      bfr[nf] = *(const v16bf*)q;
    }
    #pragma unroll
    for (int mf = 0; mf < 2; ++mf)
      #pragma unroll
      for (int nf = 0; nf < 2; ++nf)
        acc[mf][nf] = __builtin_amdgcn_wmma_f32_16x16x32_bf16(
            false, afr[mf], false, bfr[nf], (short)0, acc[mf][nf], false, false);
  }

  // exchange gate pre-activations through LDS
  #pragma unroll
  for (int mf = 0; mf < 2; ++mf)
    #pragma unroll
    for (int nf = 0; nf < 2; ++nf) {
      const int ml = nf * 16 + l16;
      #pragma unroll
      for (int r = 0; r < 8; ++r)
        lds_g[gate][wm * 32 + mf * 16 + half * 8 + r][ml] = acc[mf][nf][r];
    }
  __syncthreads();

  // fused cell update: 64x32 = 2048 elements, 8 per thread, branch-free
  #pragma unroll
  for (int e = 0; e < 8; ++e) {
    const int el = tid + e * 256;
    const int rl = el >> 5;           // 0..63
    const int ml = el & 31;           // 0..31
    const int idx = (rowb + rl) * kH + m0 + ml;
    const float ig = fsig(lds_g[0][rl][ml]);
    const float fg = fsig(lds_g[1][rl][ml]);
    const float gg = ftanh(lds_g[2][rl][ml]);
    const float og = fsig(lds_g[3][rl][ml]);
    const float cn = fg * c[idx] + ig * gg;
    c[idx] = cn;
    h_out[idx] = (__bf16)(og * ftanh(cn));
  }
}

// ---------------- batched decoder head over all T*B rows ----------------
// out[row] = act(ys[row] @ Wlin^T + blin), one wave per row.

__global__ __launch_bounds__(256) void head_all(const __bf16* __restrict__ ys,   // [T*B,1024]
                                                const float* __restrict__ Wlin,  // [4,1024]
                                                const float* __restrict__ blin,  // [4]
                                                float* __restrict__ out) {       // [T*B,4]
  const int lane = threadIdx.x & 31;
  const int wave = threadIdx.x >> 5;
  const int row  = blockIdx.x * 8 + wave;
  const __bf16* hp = ys + (size_t)row * kH;
  float p0 = 0.f, p1 = 0.f, p2 = 0.f, p3 = 0.f;
  // per-lane contiguous 32-element chunk (64B) for clean b128 vectorization
  #pragma unroll 8
  for (int j = 0; j < 32; ++j) {
    const int m = lane * 32 + j;
    const float hv = (float)hp[m];
    p0 += hv * Wlin[m];
    p1 += hv * Wlin[kH + m];
    p2 += hv * Wlin[2 * kH + m];
    p3 += hv * Wlin[3 * kH + m];
  }
  #pragma unroll
  for (int s = 16; s > 0; s >>= 1) {
    p0 += __shfl_xor(p0, s, 32);
    p1 += __shfl_xor(p1, s, 32);
    p2 += __shfl_xor(p2, s, 32);
    p3 += __shfl_xor(p3, s, 32);
  }
  if (lane == 0) {
    out[row * 4 + 0] = ftanh(p0 + blin[0]);
    out[row * 4 + 1] = ftanh(p1 + blin[1]);
    out[row * 4 + 2] = fsig(p2 + blin[2]);
    out[row * 4 + 3] = fsig(p3 + blin[3]);
  }
}

// ---------------- per-step head (fallback when ws too small for ys) ----------------

__global__ __launch_bounds__(128) void out_linear(const __bf16* __restrict__ h_bf, // [256,1024]
                                                  const float* __restrict__ Wlin,
                                                  const float* __restrict__ blin,
                                                  float* __restrict__ out) {       // [256,4]
  __shared__ float red[128];
  const int b = blockIdx.x;
  const int tid = threadIdx.x;
  float p[4] = {0.f, 0.f, 0.f, 0.f};
  for (int m = tid; m < kH; m += 128) {
    const float hv = (float)h_bf[b * kH + m];
    #pragma unroll
    for (int o = 0; o < 4; ++o) p[o] += hv * Wlin[o * kH + m];
  }
  #pragma unroll
  for (int o = 0; o < 4; ++o) {
    red[tid] = p[o];
    __syncthreads();
    for (int s = 64; s > 0; s >>= 1) {
      if (tid < s) red[tid] += red[tid + s];
      __syncthreads();
    }
    if (tid == 0) {
      const float v = red[0] + blin[o];
      out[b * 4 + o] = (o < 2) ? ftanh(v) : fsig(v);
    }
    __syncthreads();
  }
}

// ---------------- host driver ----------------

extern "C" void kernel_launch(void* const* d_in, const int* in_sizes, int n_in,
                              void* d_out, int out_size, void* d_ws, size_t ws_size,
                              hipStream_t stream) {
  const float* x_en   = (const float*)d_in[0];   // [128,256,4]
  const float* x_de   = (const float*)d_in[1];   // [128,256,4]
  const float* h0     = (const float*)d_in[2];   // [256,1024]
  const float* c0     = (const float*)d_in[3];   // [256,1024]
  const float* Wih_en = (const float*)d_in[4];   // [4096,4]
  const float* Whh_en = (const float*)d_in[5];   // [4096,1024]
  const float* b_en   = (const float*)d_in[6];   // [4096]
  const float* Wih_de = (const float*)d_in[7];
  const float* Whh_de = (const float*)d_in[8];
  const float* b_de   = (const float*)d_in[9];
  const float* Wlin   = (const float*)d_in[10];  // [4,1024]
  const float* blin   = (const float*)d_in[11];  // [4]
  float* out = (float*)d_out;                    // [128*256, 4]

  char* ws = (char*)d_ws;
  __bf16* wEn = (__bf16*)(ws);                                    // 8 MB
  __bf16* wDe = (__bf16*)(ws + 8u * 1024 * 1024);                 // 8 MB
  __bf16* hA  = (__bf16*)(ws + 16u * 1024 * 1024);                // 512 KB
  __bf16* hB  = (__bf16*)(ws + 16u * 1024 * 1024 + 512u * 1024);  // 512 KB
  float*  c   = (float*) (ws + 17u * 1024 * 1024);                // 1 MB
  __bf16* ys  = (__bf16*)(ws + 18u * 1024 * 1024);                // 64 MB (optional)
  const size_t need_ys = 18u * 1024 * 1024 + (size_t)kTDE * kB * kH * sizeof(__bf16);
  const bool batched_head = (ws_size >= need_ys);

  const int nW = kG * kH;
  cvt_f32_bf16<<<(nW + 255) / 256, 256, 0, stream>>>(Whh_en, wEn, nW);
  cvt_f32_bf16<<<(nW + 255) / 256, 256, 0, stream>>>(Whh_de, wDe, nW);
  init_state<<<(kB * kH) / 256, 256, 0, stream>>>(h0, c0, hA, c);

  const dim3 step_grid(kH / 32, kB / 64);   // (32, 4) = 128 blocks
  __bf16* hbuf[2] = { hA, hB };
  int cur = 0;

  for (int t = 0; t < kTEN; ++t) {
    lstm_step<<<step_grid, 256, 0, stream>>>(wEn, Wih_en, b_en,
                                             x_en + (size_t)t * kB * kIN,
                                             hbuf[cur], c, hbuf[cur ^ 1]);
    cur ^= 1;
  }

  if (batched_head) {
    // decoder h_t lives in its own ys slice (no aliasing); one head kernel at the end
    const __bf16* h_prev = hbuf[cur];
    for (int t = 0; t < kTDE; ++t) {
      __bf16* h_t = ys + (size_t)t * kB * kH;
      lstm_step<<<step_grid, 256, 0, stream>>>(wDe, Wih_de, b_de,
                                               x_de + (size_t)t * kB * kIN,
                                               h_prev, c, h_t);
      h_prev = h_t;
    }
    head_all<<<(kTDE * kB) / 8, 256, 0, stream>>>(ys, Wlin, blin, out);
  } else {
    for (int t = 0; t < kTDE; ++t) {
      lstm_step<<<step_grid, 256, 0, stream>>>(wDe, Wih_de, b_de,
                                               x_de + (size_t)t * kB * kIN,
                                               hbuf[cur], c, hbuf[cur ^ 1]);
      cur ^= 1;
      out_linear<<<kB, 128, 0, stream>>>(hbuf[cur], Wlin, blin, out + (size_t)t * kB * 4);
    }
  }
}
